// AttentionX_19378892439928
// MI455X (gfx1250) — compile-verified, matrix operand order checked by reading, not run on hardware
//
#include <hip/hip_runtime.h>
#include <hip/hip_bf16.h>
#include <math.h>

#define NTOK 2048
#define CQD  256
#define HN   8
#define CH   32
#define HC   256   // H * C_hid

typedef __bf16 bf16;
typedef __attribute__((ext_vector_type(16))) __bf16 v16bf;
typedef __attribute__((ext_vector_type(8)))  __bf16 v8bf;
typedef __attribute__((ext_vector_type(8)))  float  v8f;
typedef __attribute__((ext_vector_type(4))) unsigned int u32x4;
typedef __attribute__((ext_vector_type(8))) int          i32x8;
typedef __attribute__((ext_vector_type(4))) int          i32x4;

static __device__ __forceinline__ v8f wmma_bf16(v16bf a, v16bf b, v8f c) {
  // 8 args: (neg_a, A, neg_b, B, c_mod, C, reuse_a, reuse_b)
  return __builtin_amdgcn_wmma_f32_16x16x32_bf16(false, a, false, b, (short)0, c, false, false);
}

// A/B fragment loader from bf16 memory (global or LDS), ISA 7.12.2 16-bit layout:
// lane L -> row (L&15); elements [ko..ko+7] and [ko+16..ko+23], ko = (L&16)?8:0.
static __device__ __forceinline__ v16bf ldfrag_bf16(const bf16* p0, int row_stride) {
  const int l  = threadIdx.x & 31;
  const int rr = l & 15;
  const int ko = (l & 16) ? 8 : 0;
  const bf16* p = p0 + rr * row_stride + ko;
  union { v16bf v; v8bf h[2]; } u;
  u.h[0] = *(const v8bf*)(p);
  u.h[1] = *(const v8bf*)(p + 16);
  return u.v;
}

// A fragment from f32 memory (convert to bf16 on the fly), elements contiguous along K.
static __device__ __forceinline__ v16bf ldfrag_f32_row(const float* p0, int row_stride) {
  const int l  = threadIdx.x & 31;
  const int rr = l & 15;
  const int ko = (l & 16) ? 8 : 0;
  const float* p = p0 + rr * row_stride + ko;
  v16bf f;
#pragma unroll
  for (int i = 0; i < 8; ++i) { f[i] = (bf16)p[i]; f[i + 8] = (bf16)p[i + 16]; }
  return f;
}

// B fragment from f32 weight: element (k,n) at p0[k*kstride + n]; lane column n = L&15.
static __device__ __forceinline__ v16bf ldfrag_f32_col(const float* p0, int kstride) {
  const int l  = threadIdx.x & 31;
  const int n  = l & 15;
  const int ko = (l & 16) ? 8 : 0;
  const float* p = p0 + n + ko * kstride;
  v16bf f;
#pragma unroll
  for (int i = 0; i < 8; ++i) { f[i] = (bf16)p[i * kstride]; f[i + 8] = (bf16)p[(16 + i) * kstride]; }
  return f;
}

// ---------------------------------------------------------------------------
// TDM: issue an async 2-D tile load (16 rows x 32 f32 cols, row stride NTOK)
// from global bias into LDS.  D# built per ISA ch.8 (data_size=4B, type=2).
// All fields are wave-uniform (block-index/loop derived) -> SGPR groups.
// This toolchain exposes the 6-arg builtin: (g0, g1, g2, g3, g4, cpol).
// ---------------------------------------------------------------------------
static __device__ __forceinline__ void tdm_load_bias_tile(unsigned int lds_addr,
                                                          unsigned long long gaddr) {
  u32x4 g0;
  g0[0] = 1u;                                        // count=1 valid, user D#
  g0[1] = lds_addr;                                  // LDS byte address
  g0[2] = (unsigned int)gaddr;                       // global_addr[31:0]
  g0[3] = (unsigned int)(gaddr >> 32) | 0x80000000u; // global_addr[56:32] | type=2
  i32x8 g1;
  g1[0] = (int)(2u << 16);       // workgroup_mask=0, data_size=2 (4B), no flags
  g1[1] = (int)(2048u << 16);    // tensor_dim0[15:0]=2048 in [31:16]
  g1[2] = (int)(16u << 16);      // tensor_dim0[31:16]=0, tensor_dim1[15:0]=16
  g1[3] = (int)(32u << 16);      // tensor_dim1[31:16]=0, tile_dim0=32
  g1[4] = 16;                    // tile_dim1=16, tile_dim2=0 (2-D)
  g1[5] = 2048;                  // tensor_dim0_stride[31:0] = 2048 elements
  g1[6] = 0;                     // stride hi / tensor_dim1_stride lo
  g1[7] = 0;
  i32x4 z4 = {0, 0, 0, 0};
  i32x8 z8 = {0, 0, 0, 0, 0, 0, 0, 0};
  __builtin_amdgcn_tensor_load_to_lds(g0, g1, z4, z4, z8, 0);
}

// ---------------------------------------------------------------------------
// Kernel 1: fused QKVG projections.  grid (N/16, HC/16, 4), block 32 (1 wave)
// ---------------------------------------------------------------------------
__global__ __launch_bounds__(32) void proj_kernel(
    const float* __restrict__ qx, const float* __restrict__ kvx,
    const float* __restrict__ Wq, const float* __restrict__ Wk,
    const float* __restrict__ Wv, const float* __restrict__ Wg,
    bf16* __restrict__ Qb, bf16* __restrict__ Kb, bf16* __restrict__ Vt,
    float* __restrict__ G) {
  const int tn = blockIdx.x;
  const int tc = blockIdx.y;
  const int w  = blockIdx.z;
  const float* x = (w == 1 || w == 2) ? kvx : qx;
  const float* W = (w == 0) ? Wq : (w == 1) ? Wk : (w == 2) ? Wv : Wg;

  v8f acc = {};
#pragma unroll
  for (int k = 0; k < CQD; k += 32) {
    v16bf a = ldfrag_f32_row(x + tn * 16 * CQD + k, CQD);
    v16bf b = ldfrag_f32_col(W + k * HC + tc * 16, HC);
    acc = wmma_bf16(a, b, acc);
  }

  const int l = threadIdx.x & 31;
  const int col = l & 15;
  const int rbase = (l & 16) ? 8 : 0;
  const int cg = tc * 16 + col;
  const int h = cg >> 5, c = cg & 31;
#pragma unroll
  for (int r = 0; r < 8; ++r) {
    const int m = tn * 16 + rbase + r;
    const float v = acc[r];
    if (w == 0)      Qb[(h * NTOK + m) * CH + c] = (bf16)(v * 0.17677669529663689f); // 1/sqrt(32)
    else if (w == 1) Kb[(h * NTOK + m) * CH + c] = (bf16)v;
    else if (w == 2) Vt[(h * CH + c) * NTOK + m] = (bf16)v;
    else             G[m * HC + cg] = 1.0f / (1.0f + __expf(-v));
  }
}

// ---------------------------------------------------------------------------
// Kernel 2: streaming flash attention + bias + gating.
// grid (N/16, H), block 32 (1 wave) so TDM descriptors are SALU-uniform.
// Bias tiles double-buffered in LDS via tensor_load_to_lds / s_wait_tensorcnt.
// ---------------------------------------------------------------------------
__global__ __launch_bounds__(32) void attn_kernel(
    const bf16* __restrict__ Qb, const bf16* __restrict__ Kb,
    const bf16* __restrict__ Vt, const float* __restrict__ bias,
    const float* __restrict__ G, bf16* __restrict__ Og) {
  __shared__ float bias_tile[2][16 * 32];  // TDM double buffer (4 KB)
  __shared__ bf16  pbuf[16 * 32];          // P re-swizzle buffer (1 KB)

  const int l     = threadIdx.x & 31;
  const int col   = l & 15;
  const int rbase = (l & 16) ? 8 : 0;
  const int h     = blockIdx.y;
  const int q0    = blockIdx.x * 16;

  const unsigned int lds_b0 = (unsigned int)(uintptr_t)(&bias_tile[0][0]);
  const unsigned int lds_b1 = (unsigned int)(uintptr_t)(&bias_tile[1][0]);
  const unsigned long long gbase =
      (unsigned long long)(uintptr_t)(bias + ((size_t)h * NTOK + q0) * NTOK);

  v16bf aq = ldfrag_bf16(Qb + (h * NTOK + q0) * CH, CH);
  v8f o0 = {}; v8f o1 = {};
  float mrow[8], lpart[8];
#pragma unroll
  for (int r = 0; r < 8; ++r) { mrow[r] = -3.0e38f; lpart[r] = 0.0f; }

  // prologue: start DMA of bias tile 0
  tdm_load_bias_tile(lds_b0, gbase);

  for (int j = 0; j < NTOK; j += 32) {
    const int buf = (j >> 5) & 1;
    if (j + 32 < NTOK) {  // kick DMA for next tile, then wait for current one
      tdm_load_bias_tile(buf ? lds_b0 : lds_b1,
                         gbase + (unsigned long long)(j + 32) * 4ull);
      __builtin_amdgcn_s_wait_tensorcnt(1);
    } else {
      __builtin_amdgcn_s_wait_tensorcnt(0);
    }
    asm volatile("" ::: "memory");

    // scores for keys [j, j+32): two 16x16 tiles
    v16bf bk0 = ldfrag_bf16(Kb + (h * NTOK + j) * CH, CH);
    v16bf bk1 = ldfrag_bf16(Kb + (h * NTOK + j + 16) * CH, CH);
    v8f s0 = {}; v8f s1 = {};
    s0 = wmma_bf16(aq, bk0, s0);
    s1 = wmma_bf16(aq, bk1, s1);

    const float* bt = bias_tile[buf];
    float p0v[8], p1v[8];
#pragma unroll
    for (int r = 0; r < 8; ++r) {   // bias add from LDS tile
      p0v[r] = s0[r] + bt[(rbase + r) * 32 + col];
      p1v[r] = s1[r] + bt[(rbase + r) * 32 + col + 16];
    }

#pragma unroll
    for (int r = 0; r < 8; ++r) {   // online softmax: only row-max reduced here
      float tmax = fmaxf(p0v[r], p1v[r]);
#pragma unroll
      for (int d = 1; d < 16; d <<= 1) tmax = fmaxf(tmax, __shfl_xor(tmax, d, 32));
      const float mnew = fmaxf(mrow[r], tmax);
      const float corr = __expf(mrow[r] - mnew);
      const float e0 = __expf(p0v[r] - mnew);
      const float e1 = __expf(p1v[r] - mnew);
      lpart[r] = lpart[r] * corr + e0 + e1;   // per-lane partial row sum
      mrow[r] = mnew;
      o0[r] *= corr; o1[r] *= corr;
      pbuf[(rbase + r) * 32 + col]      = (bf16)e0;  // C-layout -> row-major
      pbuf[(rbase + r) * 32 + col + 16] = (bf16)e1;
    }

    // re-load P in A-fragment layout (same-wave LDS ops are in-order)
    v16bf ap  = ldfrag_bf16(pbuf, 32);
    v16bf bv0 = ldfrag_bf16(Vt + (size_t)(h * CH) * NTOK + j, NTOK);
    v16bf bv1 = ldfrag_bf16(Vt + (size_t)(h * CH + 16) * NTOK + j, NTOK);
    o0 = wmma_bf16(ap, bv0, o0);
    o1 = wmma_bf16(ap, bv1, o1);
  }

#pragma unroll
  for (int r = 0; r < 8; ++r) {
    float ls = lpart[r];            // one final cross-lane row-sum reduction
#pragma unroll
    for (int d = 1; d < 16; d <<= 1) ls += __shfl_xor(ls, d, 32);
    const int m = q0 + rbase + r;
    const float inv = 1.0f / ls;
    const int cb = h * CH;
    const float g0 = G[m * HC + cb + col];
    const float g1 = G[m * HC + cb + 16 + col];
    Og[m * HC + cb + col]      = (bf16)(o0[r] * inv * g0);
    Og[m * HC + cb + 16 + col] = (bf16)(o1[r] * inv * g1);
  }
}

// ---------------------------------------------------------------------------
// Kernel 3: output projection out = Og @ Wo.  grid (N/16, CQ/16), block 32.
// ---------------------------------------------------------------------------
__global__ __launch_bounds__(32) void oproj_kernel(
    const bf16* __restrict__ Og, const float* __restrict__ Wo,
    float* __restrict__ out) {
  const int tn = blockIdx.x, tc = blockIdx.y;
  v8f acc = {};
#pragma unroll
  for (int k = 0; k < HC; k += 32) {
    v16bf a = ldfrag_bf16(Og + tn * 16 * HC + k, HC);
    v16bf b = ldfrag_f32_col(Wo + k * CQD + tc * 16, CQD);
    acc = wmma_bf16(a, b, acc);
  }
  const int l = threadIdx.x & 31;
  const int col = l & 15;
  const int rbase = (l & 16) ? 8 : 0;
#pragma unroll
  for (int r = 0; r < 8; ++r) {
    out[(tn * 16 + rbase + r) * CQD + tc * 16 + col] = acc[r];
  }
}

extern "C" void kernel_launch(void* const* d_in, const int* in_sizes, int n_in,
                              void* d_out, int out_size, void* d_ws, size_t ws_size,
                              hipStream_t stream) {
  const float* qx   = (const float*)d_in[0];
  const float* kvx  = (const float*)d_in[1];
  const float* bias = (const float*)d_in[2];
  const float* Wq   = (const float*)d_in[3];
  const float* Wk   = (const float*)d_in[4];
  const float* Wv   = (const float*)d_in[5];
  const float* Wg   = (const float*)d_in[6];
  const float* Wo   = (const float*)d_in[7];

  char* ws = (char*)d_ws;                 // layout: 6 MB total
  bf16*  Qb = (bf16*)(ws);                // 1 MB
  bf16*  Kb = (bf16*)(ws + (1u << 20));   // 1 MB
  bf16*  Vt = (bf16*)(ws + (2u << 20));   // 1 MB
  float* G  = (float*)(ws + (3u << 20));  // 2 MB
  bf16*  Og = (bf16*)(ws + (5u << 20));   // 1 MB
  float* out = (float*)d_out;

  proj_kernel<<<dim3(NTOK / 16, HC / 16, 4), 32, 0, stream>>>(qx, kvx, Wq, Wk, Wv, Wg, Qb, Kb, Vt, G);
  attn_kernel<<<dim3(NTOK / 16, HN), 32, 0, stream>>>(Qb, Kb, Vt, bias, G, Og);
  oproj_kernel<<<dim3(NTOK / 16, CQD / 16), 32, 0, stream>>>(Og, Wo, out);
}
